// MultiHeadAttention_62809601737023
// MI455X (gfx1250) — matrix operand order repaired and checked by reading, not compiled, not run on hardware
//
#include <hip/hip_runtime.h>
#include <hip/hip_bf16.h>

// ---------------------------------------------------------------------------
// Shapes (fixed by the reference)
// ---------------------------------------------------------------------------
#define BATCH 2
#define SEQ   384
#define DMODEL 512
#define NHEAD 8
#define DKH   64

typedef __attribute__((ext_vector_type(16))) __bf16 v16bf;
typedef __attribute__((ext_vector_type(4)))  __bf16 v4bf;
typedef __attribute__((ext_vector_type(8)))  float  v8f;
typedef __attribute__((ext_vector_type(4)))  float  v4f;
typedef __attribute__((ext_vector_type(2)))  float  v2f;

// ---------------------------------------------------------------------------
// Generic LDS-staged WMMA GEMM:  C[M,N] = A[M,K] * op(B) (+ bias)
//   BT=true : B stored as [N][K] rows (NT gemm: y = A * B^T) — projections,
//             Q*K^T.
//   BT=false: B stored as [K][N]      (NN gemm)             — attn*V.
// Batched over blockIdx.z decomposed as (zb=z/nh, zh=z%nh) so the (b,h)
// head-view offsets are expressible. M,N multiples of 64; K multiple of 32.
// ---------------------------------------------------------------------------
template <bool BT>
__global__ __launch_bounds__(256) void wmma_gemm_k(
    const float* __restrict__ A, int lda, long long sA1, long long sA2,
    const float* __restrict__ B, int ldb, long long sB1, long long sB2,
    const float* __restrict__ bias,
    float* __restrict__ C, int ldc, long long sC1, long long sC2,
    int K, int nh)
{
    const int z  = blockIdx.z;
    const int zb = z / nh, zh = z % nh;
    A += (size_t)zb * sA2 + (size_t)zh * sA1;
    B += (size_t)zb * sB2 + (size_t)zh * sB1;
    C += (size_t)zb * sC2 + (size_t)zh * sC1;

    const int m0 = blockIdx.y * 64;
    const int n0 = blockIdx.x * 64;

    __shared__ __bf16 As[64][36];   // [m][k]; 72B row pitch keeps v4bf aligned
    __shared__ __bf16 Bs[64][36];   // [n][k]

    const int t    = threadIdx.x;
    const int w    = t >> 5;        // wave id 0..7
    const int l    = t & 31;        // lane
    const int half = l >> 4;        // lane half (ISA bf16 fragment layout)
    const int l16  = l & 15;
    const int mstrip = (w & 3) * 16;      // 4 m-strips of 16
    const int nbase  = (w >> 2) * 32;     // 2 n-halves of 32 (2 subtiles each)

    v8f acc0 = {}; v8f acc1 = {};

    for (int k0 = 0; k0 < K; k0 += 32) {
        // ---- stage A tile 64x32: b128 loads, packed b64 LDS stores ------
        #pragma unroll
        for (int i = 0; i < 2; ++i) {
            int idx = t + i * 256;           // float4 index 0..511
            int r = idx >> 3, c4 = idx & 7;  // row, 16B chunk
            v4f x = *(const v4f*)(A + (size_t)(m0 + r) * lda + k0 + c4 * 4);
            v4bf y;
            y[0] = (__bf16)x[0]; y[1] = (__bf16)x[1];
            y[2] = (__bf16)x[2]; y[3] = (__bf16)x[3];
            *(v4bf*)&As[r][c4 * 4] = y;
        }
        // ---- stage B tile into [n][k] -----------------------------------
        if (BT) {
            #pragma unroll
            for (int i = 0; i < 2; ++i) {
                int idx = t + i * 256;
                int r = idx >> 3, c4 = idx & 7;
                v4f x = *(const v4f*)(B + (size_t)(n0 + r) * ldb + k0 + c4 * 4);
                v4bf y;
                y[0] = (__bf16)x[0]; y[1] = (__bf16)x[1];
                y[2] = (__bf16)x[2]; y[3] = (__bf16)x[3];
                *(v4bf*)&Bs[r][c4 * 4] = y;
            }
        } else {
            // transpose path (small attn*V GEMM only): coalesced along n
            #pragma unroll
            for (int i = 0; i < 8; ++i) {
                int idx = t + i * 256;
                int kk = idx >> 6, nn = idx & 63;
                Bs[nn][kk] = (__bf16)B[(size_t)(k0 + kk) * ldb + n0 + nn];
            }
        }
        __syncthreads();

        // ---- fragments per ISA 7.12.2 bf16 layouts ----------------------
        // A 16x32: lanes0-15 elems0-7=K0..7, elems8-15=K16..23; lanes16-31 +8
    // B 32x16: lane holds column N=l16; K order same as A
        v16bf af, bf0, bf1;
        #pragma unroll
        for (int j = 0; j < 16; ++j) {
            int ka = (j < 8) ? (half * 8 + j) : (16 + half * 8 + (j - 8));
            af[j]  = As[mstrip + l16][ka];
            int kb = ka;
            bf0[j] = Bs[nbase + l16][kb];
            bf1[j] = Bs[nbase + 16 + l16][kb];
        }
        acc0 = __builtin_amdgcn_wmma_f32_16x16x32_bf16(
            false, af, false, bf0, (short)0, acc0, false, false);
        acc1 = __builtin_amdgcn_wmma_f32_16x16x32_bf16(
            false, af, false, bf1, (short)0, acc1, false, false);
        __syncthreads();
    }

    // ---- epilogue: C/D layout — VGPR r, lane: M = r + half*8, N = l16 ----
    #pragma unroll
    for (int r = 0; r < 8; ++r) {
        int row = m0 + mstrip + half * 8 + r;
        int c0  = n0 + nbase + l16;
        float v0 = acc0[r], v1 = acc1[r];
        if (bias) { v0 += bias[c0]; v1 += bias[c0 + 16]; }
        C[(size_t)row * ldc + c0]      = v0;
        C[(size_t)row * ldc + c0 + 16] = v1;
    }
}

// ---------------------------------------------------------------------------
// Bias-K contraction: scores[b,h,i,j] += sum_d q[b,i,h,d] * ebk[b,j,i,h,d]
// One block per (b,i); wave w owns head h=w. One b64 load + 2 FMA per lane
// per j; wave covers its head's 256B slice. Streams 604 MB coalesced.
// ---------------------------------------------------------------------------
__global__ __launch_bounds__(256) void bias_k_add_k(
    const float* __restrict__ qp, const float* __restrict__ ebk,
    float* __restrict__ scores)
{
    const int bi = blockIdx.x;            // b*SEQ + i
    const int b = bi / SEQ, i = bi % SEQ;
    __shared__ float qs[DMODEL];
    const int t = threadIdx.x;
    qs[t]       = qp[(size_t)bi * DMODEL + t];
    qs[t + 256] = qp[(size_t)bi * DMODEL + t + 256];
    __syncthreads();

    const int h = t >> 5, l = t & 31;
    const float qa = qs[h * DKH + 2 * l];
    const float qb = qs[h * DKH + 2 * l + 1];
    const size_t jstride = (size_t)SEQ * DMODEL;
    const float* base = ebk + ((size_t)b * SEQ * SEQ + i) * DMODEL + h * DKH;
    float* srow = scores + (((size_t)b * NHEAD + h) * SEQ + i) * SEQ;

    for (int j = 0; j < SEQ; ++j) {
        const float* row = base + (size_t)j * jstride;
        if (j + 8 < SEQ)
            __builtin_prefetch(row + 8 * jstride + l * 2, 0, 0);
        v2f r2 = *(const v2f*)(row + 2 * l);
        float p = qa * r2[0] + qb * r2[1];
        #pragma unroll
        for (int off = 16; off > 0; off >>= 1)
            p += __shfl_down(p, off, 32);
        if (l == 0) srow[j] += p;
    }
}

// ---------------------------------------------------------------------------
// Masked softmax over j, in place. Reference masks on the QUERY index i
// (mask[:,None,:,None]) -> masked rows are exactly uniform 1/SEQ.
// Scale 1/sqrt(DK)=0.125 applied here.
// ---------------------------------------------------------------------------
__global__ __launch_bounds__(128) void softmax_mask_k(
    float* __restrict__ scores, const int* __restrict__ mask)
{
    const int r = blockIdx.x;             // (b*NHEAD+h)*SEQ + i
    const int i = r % SEQ;
    const int b = r / (NHEAD * SEQ);
    float* row = scores + (size_t)r * SEQ;
    const int t = threadIdx.x;
    __shared__ float red[128];

    if (mask[b * SEQ + i] == 1) {
        const float u = 1.0f / (float)SEQ;
        for (int j = t; j < SEQ; j += 128) row[j] = u;
        return;
    }
    float v[3];
    float mx = -3.0e38f;
    #pragma unroll
    for (int c = 0; c < 3; ++c) {
        v[c] = row[t + c * 128] * 0.125f;
        mx = fmaxf(mx, v[c]);
    }
    red[t] = mx; __syncthreads();
    for (int off = 64; off > 0; off >>= 1) {
        if (t < off) red[t] = fmaxf(red[t], red[t + off]);
        __syncthreads();
    }
    mx = red[0]; __syncthreads();
    float s = 0.f;
    #pragma unroll
    for (int c = 0; c < 3; ++c) { v[c] = __expf(v[c] - mx); s += v[c]; }
    red[t] = s; __syncthreads();
    for (int off = 64; off > 0; off >>= 1) {
        if (t < off) red[t] += red[t + off];
        __syncthreads();
    }
    const float inv = 1.0f / red[0];
    #pragma unroll
    for (int c = 0; c < 3; ++c) row[t + c * 128] = v[c] * inv;
}

// ---------------------------------------------------------------------------
// Bias-V contraction: out1[b,i,h,d] += sum_j attn[b,h,i,j] * ebv[b,j,i,h,d]
// One block per (b,i), 128 threads, one b128 load + 4 FMA per lane per j
// (block covers the full 2KB row). Streams 604 MB coalesced.
// ---------------------------------------------------------------------------
__global__ __launch_bounds__(128) void bias_v_add_k(
    const float* __restrict__ attn, const float* __restrict__ ebv,
    float* __restrict__ out1)
{
    const int bi = blockIdx.x;
    const int b = bi / SEQ, i = bi % SEQ;
    const int t = threadIdx.x;            // float4 index: d = 4t
    const int h = t >> 4;                 // (4t)/64
    __shared__ float aw[NHEAD][128];
    const size_t jstride = (size_t)SEQ * DMODEL;
    const float* base = ebv + ((size_t)b * SEQ * SEQ + i) * DMODEL;
    v4f acc = {};

    for (int jc = 0; jc < SEQ; jc += 128) {
        __syncthreads();
        #pragma unroll
        for (int rep = 0; rep < 8; ++rep) {   // stage 8x128 attn weights
            int idx = t + rep * 128;
            int hh = idx >> 7, jj = idx & 127;
            aw[hh][jj] =
                attn[(((size_t)b * NHEAD + hh) * SEQ + i) * SEQ + jc + jj];
        }
        __syncthreads();
        for (int jj = 0; jj < 128; ++jj) {
            const float* row = base + (size_t)(jc + jj) * jstride;
            if (jc + jj + 4 < SEQ)
                __builtin_prefetch(row + 4 * jstride + t * 4, 0, 0);
            v4f rv = *(const v4f*)(row + t * 4);
            acc += rv * aw[h][jj];
        }
    }
    v4f* op = (v4f*)(out1 + (size_t)bi * DMODEL + t * 4);
    *op += acc;
}

// ---------------------------------------------------------------------------
// Launch: 9 stream-ordered kernels.
// Workspace (floats): qp | kp | vp | out1 | scores  = ~15.7 MB total.
// ---------------------------------------------------------------------------
extern "C" void kernel_launch(void* const* d_in, const int* in_sizes, int n_in,
                              void* d_out, int out_size, void* d_ws,
                              size_t ws_size, hipStream_t stream)
{
    const float* q   = (const float*)d_in[0];
    const float* k   = (const float*)d_in[1];
    const float* v   = (const float*)d_in[2];
    const float* ebk = (const float*)d_in[4];
    const float* ebv = (const float*)d_in[5];
    const int*   msk = (const int*)d_in[6];
    const float* Wq  = (const float*)d_in[7];
    const float* bq  = (const float*)d_in[8];
    const float* Wk  = (const float*)d_in[9];
    const float* bk  = (const float*)d_in[10];
    const float* Wv  = (const float*)d_in[11];
    const float* bv  = (const float*)d_in[12];
    const float* Wo  = (const float*)d_in[13];
    const float* bo  = (const float*)d_in[14];
    float* out = (float*)d_out;

    const int M  = BATCH * SEQ;               // 768
    const long long SS  = (long long)SEQ * SEQ;
    const long long SD  = (long long)SEQ * DMODEL;

    float* ws = (float*)d_ws;
    float* qp = ws;                           // [768,512] = [b,i,h,d]
    float* kp = qp + (size_t)M * DMODEL;
    float* vp = kp + (size_t)M * DMODEL;
    float* o1 = vp + (size_t)M * DMODEL;      // attn output, [b,i,h,d]
    float* sc = o1 + (size_t)M * DMODEL;      // scores/attn [b,h,i,j]

    dim3 blk(256);

    // Q/K/V projections: [768x512] = X[768x512] * W^T + b  (NT)
    dim3 gproj(DMODEL / 64, M / 64, 1);
    wmma_gemm_k<true><<<gproj, blk, 0, stream>>>(
        q, DMODEL, 0, 0, Wq, DMODEL, 0, 0, bq, qp, DMODEL, 0, 0, DMODEL, 1);
    wmma_gemm_k<true><<<gproj, blk, 0, stream>>>(
        k, DMODEL, 0, 0, Wk, DMODEL, 0, 0, bk, kp, DMODEL, 0, 0, DMODEL, 1);
    wmma_gemm_k<true><<<gproj, blk, 0, stream>>>(
        v, DMODEL, 0, 0, Wv, DMODEL, 0, 0, bv, vp, DMODEL, 0, 0, DMODEL, 1);

    // scores[b,h] = Qh[384x64] * Kh[384x64]^T  (NT, batched over 16 (b,h))
    dim3 gsc(SEQ / 64, SEQ / 64, BATCH * NHEAD);
    wmma_gemm_k<true><<<gsc, blk, 0, stream>>>(
        qp, DMODEL, DKH, SD, kp, DMODEL, DKH, SD, nullptr,
        sc, SEQ, SS, NHEAD * SS, DKH, NHEAD);

    // += q . edge_bias_k   (streams 604 MB)
    bias_k_add_k<<<dim3(M), dim3(256), 0, stream>>>(qp, ebk, sc);

    // masked softmax (in place)
    softmax_mask_k<<<dim3(BATCH * NHEAD * SEQ), dim3(128), 0, stream>>>(sc, msk);

    // out1[b,h] = attn[384x384] * Vh[384x64]  (NN, batched)
    dim3 gav(DKH / 64, SEQ / 64, BATCH * NHEAD);
    wmma_gemm_k<false><<<gav, blk, 0, stream>>>(
        sc, SEQ, SS, NHEAD * SS, vp, DMODEL, DKH, SD, nullptr,
        o1, DMODEL, DKH, SD, SEQ, NHEAD);

    // += attn . edge_bias_v   (streams 604 MB)
    bias_v_add_k<<<dim3(M), dim3(128), 0, stream>>>(sc, ebv, o1);

    // output projection -> d_out
    wmma_gemm_k<true><<<gproj, blk, 0, stream>>>(
        o1, DMODEL, 0, 0, Wo, DMODEL, 0, 0, bo, out, DMODEL, 0, 0, DMODEL, 1);
}
